// DistributedMoELayer_29102698397784
// MI455X (gfx1250) — matrix operand order repaired and checked by reading, not compiled, not run on hardware
//
#include <hip/hip_runtime.h>
#include <hip/hip_bf16.h>

typedef __attribute__((ext_vector_type(16))) __bf16        v16bf;
typedef __attribute__((ext_vector_type(8)))  float         v8f;
typedef __attribute__((ext_vector_type(4)))  unsigned int  v4u;
typedef __attribute__((ext_vector_type(4)))  float         v4f;
typedef __attribute__((ext_vector_type(4)))  unsigned int  u32x4;
typedef __attribute__((ext_vector_type(8)))  int           i32x8;
typedef __attribute__((ext_vector_type(4)))  int           i32x4;

#define NT    16384
#define HID   1024
#define FFN   4096
#define NE    8
#define MT    128              // M tile
#define NTL   128              // N tile
#define KT    64               // K tile == two wmma k-steps (bf16 16x16x32)
#define LPITCH 72              // LDS pitch in shorts: 144B, 16B-aligned, conflict-free
#define CAP   (2*NT + NE*MT)   // 33792 padded gathered rows
#define MTILES (CAP/MT)        // 264

// ---------------- workspace layout (bytes) ----------------
static const size_t O_COUNTS = 0;                                  // 8 ints
static const size_t O_SEGOFF = 256;                                // 9 ints (segoff[8] = total)
static const size_t O_CURS   = 512;                                // 8 ints
static const size_t O_TOKE   = 1024;                               // NT*2 ints
static const size_t O_TOKW   = O_TOKE + (size_t)NT*2*4;            // NT*2 floats
static const size_t O_ROWS   = O_TOKW + (size_t)NT*2*4;            // CAP ints
static const size_t O_RW     = O_ROWS + (size_t)CAP*4;             // CAP floats
static const size_t O_W1T    = (O_RW + (size_t)CAP*4 + 255) & ~(size_t)255; // bf16 [NE][FFN][HID]
static const size_t O_W2T    = O_W1T + (size_t)NE*HID*FFN*2;       // bf16 [NE][HID][FFN]
static const size_t O_H      = O_W2T + (size_t)NE*FFN*HID*2;       // bf16 [CAP][FFN]

typedef union { v16bf v; v4u q[2]; } Frag;

__device__ __forceinline__ unsigned short f2bf(float f) {
  unsigned int u = __float_as_uint(f);
  u += 0x7fffu + ((u >> 16) & 1u);              // round-to-nearest-even
  return (unsigned short)(u >> 16);
}

__device__ __forceinline__ float gelu_tanh(float x) {
  float t = 0.7978845608028654f * (x + 0.044715f * x * x * x);
  return 0.5f * x * (1.0f + tanhf(t));
}

// ---- Tensor Data Mover: load a [tile_n rows x tile_k elems] bf16 tile to LDS,
// ---- padding 4 dwords after every 32 dwords (=> LDS pitch 72 shorts for tile_k=64).
__device__ __forceinline__ void tdm_load_tile(const void* gsrc, unsigned lds_addr,
                                              unsigned tile_k, unsigned tile_n,
                                              unsigned row_stride_elems) {
  unsigned long long ga = (unsigned long long)(uintptr_t)gsrc;
  u32x4 g0;
  g0.x = 1u;                                                   // count=1, user mode
  g0.y = lds_addr;                                             // lds_addr
  g0.z = (unsigned)ga;                                         // global_addr[31:0]
  g0.w = (unsigned)((ga >> 32) & 0x01FFFFFFu) | (2u << 30);    // global_addr[56:32], type=2
  i32x8 g1;
  g1[0] = (int)((1u << 16)      // data_size = 2 bytes
              | (1u << 20)      // pad_enable
              | (4u << 22)      // pad_interval: 32 dwords
              | (3u << 25));    // pad_amount:   4 dwords
  g1[1] = (int)((tile_k & 0xFFFFu) << 16);                          // tensor_dim0[15:0]
  g1[2] = (int)(((tile_k >> 16) & 0xFFFFu) | ((tile_n & 0xFFFFu) << 16)); // dim0 hi | dim1 lo
  g1[3] = (int)(((tile_n >> 16) & 0xFFFFu) | ((tile_k & 0xFFFFu) << 16)); // dim1 hi | tile_dim0
  g1[4] = (int)(tile_n & 0xFFFFu);                                  // tile_dim1 (tile_dim2=0)
  g1[5] = (int)row_stride_elems;                                    // tensor_dim0_stride lo
  g1[6] = 0;
  g1[7] = 0;
  i32x4 gz4 = {0, 0, 0, 0};
  i32x8 gz8 = {0, 0, 0, 0, 0, 0, 0, 0};
  __builtin_amdgcn_tensor_load_to_lds(g0, g1, gz4, gz4, gz8, 0);
}

// ---------------- init: counts=0, rows=-1, rw=0 ----------------
__global__ void k_init(int* __restrict__ counts, int* __restrict__ rows,
                       float* __restrict__ rw) {
  int i = blockIdx.x * blockDim.x + threadIdx.x;
  if (i < NE) counts[i] = 0;
  if (i < CAP) { rows[i] = -1; rw[i] = 0.0f; }
}

// ---------------- fp32 -> bf16 transpose-convert of weights ----------------
// w1[e] is [HID][FFN] -> w1t[e] is [FFN][HID]; w2[e] is [FFN][HID] -> w2t[e] is [HID][FFN]
__global__ __launch_bounds__(256)
void k_wtrans(const float* __restrict__ w1, const float* __restrict__ w2,
              unsigned short* __restrict__ w1t, unsigned short* __restrict__ w2t) {
  const int z = blockIdx.z;
  const float* src; unsigned short* dst; int K, N;
  if (z < NE) { src = w1 + (size_t)z * HID * FFN; dst = w1t + (size_t)z * HID * FFN; K = HID; N = FFN; }
  else        { src = w2 + (size_t)(z - NE) * FFN * HID; dst = w2t + (size_t)(z - NE) * FFN * HID; K = FFN; N = HID; }
  const int kb = blockIdx.x * 32, nb = blockIdx.y * 32;
  if (kb >= K || nb >= N) return;
  __shared__ float t[32][33];
  const int tx = threadIdx.x & 31, ty = threadIdx.x >> 5;   // 32 x 8
  #pragma unroll
  for (int i = 0; i < 32; i += 8)
    t[ty + i][tx] = src[(size_t)(kb + ty + i) * N + nb + tx];
  __syncthreads();
  #pragma unroll
  for (int i = 0; i < 32; i += 8)
    dst[(size_t)(nb + ty + i) * K + kb + tx] = f2bf(t[tx][ty + i]);
}

// ---------------- router: one wave per token ----------------
__global__ __launch_bounds__(256)
void k_router(const float* __restrict__ x, const float* __restrict__ wr,
              int* __restrict__ toke, float* __restrict__ tokw,
              int* __restrict__ counts) {
  const int lane  = threadIdx.x & 31;
  const int token = blockIdx.x * 8 + (threadIdx.x >> 5);
  float acc[NE] = {0, 0, 0, 0, 0, 0, 0, 0};
  for (int k = lane; k < HID; k += 32) {
    float xv = x[(size_t)token * HID + k];
    const v4f* wp = (const v4f*)(wr + (size_t)k * NE);
    v4f w0 = wp[0], w1 = wp[1];
    acc[0] += xv * w0.x; acc[1] += xv * w0.y; acc[2] += xv * w0.z; acc[3] += xv * w0.w;
    acc[4] += xv * w1.x; acc[5] += xv * w1.y; acc[6] += xv * w1.z; acc[7] += xv * w1.w;
  }
  #pragma unroll
  for (int m = 16; m > 0; m >>= 1) {
    #pragma unroll
    for (int e = 0; e < NE; ++e) acc[e] += __shfl_xor(acc[e], m, 32);
  }
  if (lane == 0) {
    int i0 = 0;
    #pragma unroll
    for (int e = 1; e < NE; ++e) if (acc[e] > acc[i0]) i0 = e;
    int i1 = (i0 == 0) ? 1 : 0;
    #pragma unroll
    for (int e = 0; e < NE; ++e) if (e != i0 && acc[e] > acc[i1]) i1 = e;
    float w0 = 1.0f / (1.0f + __expf(acc[i1] - acc[i0]));   // stable renormalized top-2
    toke[token * 2 + 0] = i0;  tokw[token * 2 + 0] = w0;
    toke[token * 2 + 1] = i1;  tokw[token * 2 + 1] = 1.0f - w0;
    atomicAdd(&counts[i0], 1);
    atomicAdd(&counts[i1], 1);
  }
}

// ---------------- scan: 128-aligned segment offsets ----------------
__global__ void k_scan(const int* __restrict__ counts, int* __restrict__ segoff,
                       int* __restrict__ curs) {
  int off = 0;
  for (int e = 0; e < NE; ++e) {
    segoff[e] = off; curs[e] = off;
    off += ((counts[e] + MT - 1) / MT) * MT;
  }
  segoff[NE] = off;
}

// ---------------- scatter tokens into expert segments ----------------
__global__ void k_scatter(const int* __restrict__ toke, const float* __restrict__ tokw,
                          int* __restrict__ curs, int* __restrict__ rows,
                          float* __restrict__ rw) {
  int t = blockIdx.x * blockDim.x + threadIdx.x;
  if (t >= NT) return;
  #pragma unroll
  for (int k = 0; k < 2; ++k) {
    int e = toke[t * 2 + k];
    int pos = atomicAdd(&curs[e], 1);
    rows[pos] = t;
    rw[pos]   = tokw[t * 2 + k];
  }
}

// ---------------- GEMM1: h = gelu(gather(x) @ w1[e]), bf16 out ----------------
__global__ __launch_bounds__(256)
void k_gemm1(const float* __restrict__ x, const unsigned short* __restrict__ w1t,
             const int* __restrict__ rows, const int* __restrict__ segoff,
             unsigned short* __restrict__ h) {
  const int mbase = blockIdx.x * MT;
  if (mbase >= segoff[NE]) return;
  int e = 0;
  #pragma unroll
  for (int i = 0; i < NE; ++i) if (mbase >= segoff[i + 1]) e = i + 1;
  const int nb = blockIdx.y * NTL;

  __shared__ unsigned short As[MT][LPITCH];
  __shared__ unsigned short Bs[NTL][LPITCH];

  const int tid  = threadIdx.x;
  const int lane = tid & 31;
  const int wv   = tid >> 5;
  const int wm   = wv & 3;                // 4 waves x 32 rows
  const int wn   = wv >> 2;               // 2 waves x 64 cols

  const int ar = tid >> 1;                // A load: row 0..127
  const int kc = (tid & 1) * 32;          //          k-chunk 0/32
  const int arow = rows[mbase + ar];      // gathered source row (-1 = pad)

  const unsigned short* wtb = w1t + (size_t)e * FFN * HID + (size_t)nb * HID;
  const unsigned bsLds = (unsigned)(uintptr_t)&Bs[0][0];
  v8f acc[2][4] = {};

  for (int kt = 0; kt < HID; kt += KT) {
    // B tile: one TDM descriptor moves 128 rows x 64 bf16, DMA-padded to pitch 72
    if (wv == 0)
      tdm_load_tile(wtb + kt, bsLds, KT, NTL, HID);
    // A tile: fp32 gather -> bf16, aligned b128 LDS stores
    {
      unsigned short tmp[8] __attribute__((aligned(16)));
      if (arow >= 0) {
        const v4f* xp = (const v4f*)(x + (size_t)arow * HID + kt + kc);
        #pragma unroll
        for (int g = 0; g < 4; ++g) {
          v4f f0 = xp[2 * g], f1 = xp[2 * g + 1];
          tmp[0] = f2bf(f0.x); tmp[1] = f2bf(f0.y); tmp[2] = f2bf(f0.z); tmp[3] = f2bf(f0.w);
          tmp[4] = f2bf(f1.x); tmp[5] = f2bf(f1.y); tmp[6] = f2bf(f1.z); tmp[7] = f2bf(f1.w);
          *(v4u*)&As[ar][kc + g * 8] = *(const v4u*)tmp;
        }
      } else {
        v4u zz = {0, 0, 0, 0};
        #pragma unroll
        for (int g = 0; g < 4; ++g) *(v4u*)&As[ar][kc + g * 8] = zz;
      }
    }
    __builtin_amdgcn_s_wait_tensorcnt(0);   // waves with no TDM pass immediately
    __syncthreads();

    #pragma unroll
    for (int kk = 0; kk < KT; kk += 32) {
      const int am  = lane & 15;            // A: lane -> M row
      const int akb = kk + (lane >> 4) * 8; // A: K chunks {akb..+7, akb+16..+23}
      const int bc  = lane & 15;            // B: lane -> N col
      const int bks = kk + (lane >> 4) * 8; // B: K chunks {bks..+7, bks+16..+23}
      Frag a[2], b[4];
      #pragma unroll
      for (int mi = 0; mi < 2; ++mi) {
        const unsigned short* p = &As[wm * 32 + mi * 16 + am][akb];
        a[mi].q[0] = *(const v4u*)p;
        a[mi].q[1] = *(const v4u*)(p + 16);
      }
      #pragma unroll
      for (int ni = 0; ni < 4; ++ni) {
        const unsigned short* p = &Bs[wn * 64 + ni * 16 + bc][bks];
        b[ni].q[0] = *(const v4u*)p;
        b[ni].q[1] = *(const v4u*)(p + 16);
      }
      #pragma unroll
      for (int mi = 0; mi < 2; ++mi)
        #pragma unroll
        for (int ni = 0; ni < 4; ++ni)
          acc[mi][ni] = __builtin_amdgcn_wmma_f32_16x16x32_bf16(
              false, a[mi].v, false, b[ni].v, (short)0, acc[mi][ni], false, false);
    }
    __syncthreads();
  }

  // epilogue: gelu -> bf16 h
  #pragma unroll
  for (int mi = 0; mi < 2; ++mi)
    #pragma unroll
    for (int ni = 0; ni < 4; ++ni)
      #pragma unroll
      for (int r = 0; r < 8; ++r) {
        int rl = wm * 32 + mi * 16 + r + ((lane >> 4) << 3);   // C: row = r + 8*(lane>=16)
        int cl = wn * 64 + ni * 16 + (lane & 15);
        h[(size_t)(mbase + rl) * FFN + nb + cl] = f2bf(gelu_tanh(acc[mi][ni][r]));
      }
}

// ---------------- GEMM2: out[token] += rw * (h @ w2[e]) ----------------
__global__ __launch_bounds__(256)
void k_gemm2(const unsigned short* __restrict__ h, const unsigned short* __restrict__ w2t,
             const int* __restrict__ rows, const float* __restrict__ rw,
             const int* __restrict__ segoff, float* __restrict__ out) {
  const int mbase = blockIdx.x * MT;
  if (mbase >= segoff[NE]) return;
  int e = 0;
  #pragma unroll
  for (int i = 0; i < NE; ++i) if (mbase >= segoff[i + 1]) e = i + 1;
  const int nb = blockIdx.y * NTL;

  __shared__ unsigned short As[MT][LPITCH];
  __shared__ unsigned short Bs[NTL][LPITCH];

  const int tid  = threadIdx.x;
  const int lane = tid & 31;
  const int wv   = tid >> 5;
  const int wm   = wv & 3;
  const int wn   = wv >> 2;

  const int ar = tid >> 1;
  const int kc = (tid & 1) * 32;

  const unsigned short* wtb = w2t + (size_t)e * HID * FFN + (size_t)nb * FFN;
  const unsigned bsLds = (unsigned)(uintptr_t)&Bs[0][0];
  const unsigned asLds = (unsigned)(uintptr_t)&As[ar][kc];
  v8f acc[2][4] = {};

  for (int kt = 0; kt < FFN; kt += KT) {
    // B tile via Tensor Data Mover
    if (wv == 0)
      tdm_load_tile(wtb + kt, bsLds, KT, NTL, FFN);
    // A tile (already bf16 in h): async copy straight to LDS, no VGPR round-trip
    {
      unsigned long long gp =
          (unsigned long long)(uintptr_t)(h + (size_t)(mbase + ar) * FFN + kt + kc);
      asm volatile(
          "global_load_async_to_lds_b128 %0, %1, off\n\t"
          "global_load_async_to_lds_b128 %0, %1, off offset:16\n\t"
          "global_load_async_to_lds_b128 %0, %1, off offset:32\n\t"
          "global_load_async_to_lds_b128 %0, %1, off offset:48"
          :: "v"(asLds), "v"(gp) : "memory");
    }
    asm volatile("s_wait_asynccnt 0x0" ::: "memory");
    __builtin_amdgcn_s_wait_tensorcnt(0);
    __syncthreads();

    #pragma unroll
    for (int kk = 0; kk < KT; kk += 32) {
      const int am  = lane & 15;
      const int akb = kk + (lane >> 4) * 8;
      const int bc  = lane & 15;
      const int bks = kk + (lane >> 4) * 8;
      Frag a[2], b[4];
      #pragma unroll
      for (int mi = 0; mi < 2; ++mi) {
        const unsigned short* p = &As[wm * 32 + mi * 16 + am][akb];
        a[mi].q[0] = *(const v4u*)p;
        a[mi].q[1] = *(const v4u*)(p + 16);
      }
      #pragma unroll
      for (int ni = 0; ni < 4; ++ni) {
        const unsigned short* p = &Bs[wn * 64 + ni * 16 + bc][bks];
        b[ni].q[0] = *(const v4u*)p;
        b[ni].q[1] = *(const v4u*)(p + 16);
      }
      #pragma unroll
      for (int mi = 0; mi < 2; ++mi)
        #pragma unroll
        for (int ni = 0; ni < 4; ++ni)
          acc[mi][ni] = __builtin_amdgcn_wmma_f32_16x16x32_bf16(
              false, a[mi].v, false, b[ni].v, (short)0, acc[mi][ni], false, false);
    }
    __syncthreads();
  }

  // epilogue: weighted atomic combine (exactly 2 commuting adds per element -> deterministic)
  #pragma unroll
  for (int mi = 0; mi < 2; ++mi)
    #pragma unroll
    for (int ni = 0; ni < 4; ++ni)
      #pragma unroll
      for (int r = 0; r < 8; ++r) {
        int rl = wm * 32 + mi * 16 + r + ((lane >> 4) << 3);
        int cl = wn * 64 + ni * 16 + (lane & 15);
        int grow = mbase + rl;
        int tok = rows[grow];
        if (tok >= 0)
          atomicAdd(out + (size_t)tok * HID + nb + cl, rw[grow] * acc[mi][ni][r]);
      }
}

// ---------------- launch ----------------
extern "C" void kernel_launch(void* const* d_in, const int* in_sizes, int n_in,
                              void* d_out, int out_size, void* d_ws, size_t ws_size,
                              hipStream_t stream) {
  const float* x  = (const float*)d_in[0];
  const float* wr = (const float*)d_in[1];
  const float* w1 = (const float*)d_in[2];
  const float* w2 = (const float*)d_in[3];
  float* out = (float*)d_out;
  char* ws = (char*)d_ws;

  int*   counts = (int*)(ws + O_COUNTS);
  int*   segoff = (int*)(ws + O_SEGOFF);
  int*   curs   = (int*)(ws + O_CURS);
  int*   toke   = (int*)(ws + O_TOKE);
  float* tokw   = (float*)(ws + O_TOKW);
  int*   rows   = (int*)(ws + O_ROWS);
  float* rwv    = (float*)(ws + O_RW);
  unsigned short* w1t = (unsigned short*)(ws + O_W1T);
  unsigned short* w2t = (unsigned short*)(ws + O_W2T);
  unsigned short* h   = (unsigned short*)(ws + O_H);

  (void)hipMemsetAsync(d_out, 0, (size_t)NT * HID * sizeof(float), stream);
  k_init<<<(CAP + 255) / 256, 256, 0, stream>>>(counts, rows, rwv);
  k_wtrans<<<dim3(FFN / 32, FFN / 32, 2 * NE), 256, 0, stream>>>(w1, w2, w1t, w2t);
  k_router<<<NT / 8, 256, 0, stream>>>(x, wr, toke, tokw, counts);
  k_scan<<<1, 1, 0, stream>>>(counts, segoff, curs);
  k_scatter<<<NT / 256, 256, 0, stream>>>(toke, tokw, curs, rows, rwv);
  k_gemm1<<<dim3(MTILES, FFN / NTL), 256, 0, stream>>>(x, w1t, rows, segoff, h);
  k_gemm2<<<dim3(MTILES, HID / NTL), 256, 0, stream>>>(h, w2t, rows, rwv, segoff, out);
}
